// ContextBlock_46368466927815
// MI455X (gfx1250) — compile-verified, matrix operand order checked by reading, not run on hardware
//
#include <hip/hip_runtime.h>
#include <stdint.h>

// ---------------------------------------------------------------------------
// CDNA5 (gfx1250) context-block: WS-conv QKV -> transposed-softmax attention
// -> WS-conv O + residual -> LayerNorm.  bf16 WMMA (f32 accum) everywhere.
//
// - Global K-permutation (K' = half*16 + j) applied identically to A and B
//   fragments => every WMMA fragment is one contiguous 32-byte LDS read
//   (2x ds_load_b128).
// - Activations kept channel-innermost (transposed) so both GEMM operands
//   are contraction-contiguous.
// - Tile fills use GLOBAL_LOAD_ASYNC_TO_LDS_B128 (ASYNCcnt) when available,
//   with a double-buffered GEMM mainloop overlapping fill and WMMA.
// ---------------------------------------------------------------------------

typedef __attribute__((ext_vector_type(16))) __bf16 v16bf;
typedef __attribute__((ext_vector_type(8)))  __bf16 bf16x8;
typedef __attribute__((ext_vector_type(8)))  float  v8f;

#define DEV __device__ __forceinline__

#if defined(__has_builtin)
#  if __has_builtin(__builtin_amdgcn_global_load_async_to_lds_b128) && \
      __has_builtin(__builtin_amdgcn_s_wait_asynccnt)
#    define CB_ASYNC 1
#  endif
#endif
#ifndef CB_ASYNC
#  define CB_ASYNC 0
#endif

#if CB_ASYNC
// Parameter types per hipcc diagnostic: v4i in addrspace(1) / addrspace(3).
typedef int v4i __attribute__((__vector_size__(4 * sizeof(int))));
typedef __attribute__((address_space(1))) v4i* gptr128;
typedef __attribute__((address_space(3))) v4i* lptr128;
#endif

namespace cb {
constexpr int B  = 8;
constexpr int TQ = 1024;
constexpr int TC = 1024;
constexpr int E  = 1024;
constexpr int C  = 512;
constexpr int H  = 8;
constexpr int DH = 128;
constexpr float EPS    = 1e-5f;
constexpr float RSCALE = 1.0f / 362.0f;   // SCALE = float(1024 // sqrt(8)) = 362.0
}

DEV v8f wmma_bf16(v16bf a, v16bf b, v8f c) {
  return __builtin_amdgcn_wmma_f32_16x16x32_bf16(false, a, false, b, (short)0, c,
                                                 false, false);
}

// One WMMA fragment = 32 contiguous bytes of LDS (K' = half*16 + j mapping).
DEV v16bf load_frag(const __bf16* p) {
  bf16x8 lo = *(const bf16x8*)p;
  bf16x8 hi = *(const bf16x8*)(p + 8);
  return __builtin_shufflevector(lo, hi, 0, 1, 2, 3, 4, 5, 6, 7,
                                 8, 9, 10, 11, 12, 13, 14, 15);
}

// 16-byte global -> LDS copy: async (ASYNCcnt-tracked, lands directly in LDS)
// when the toolchain exposes it; plain uint4 round-trip otherwise.
DEV void copy16(__bf16* dst, const __bf16* src) {
#if CB_ASYNC
  __builtin_amdgcn_global_load_async_to_lds_b128((gptr128)src, (lptr128)dst, 0, 0);
#else
  *(uint4*)dst = *(const uint4*)src;
#endif
}

DEV void copy_fence() {
#if CB_ASYNC
  __builtin_amdgcn_s_wait_asynccnt(0);
#endif
}

// ---------------------------------------------------------------------------
// Kernel 1: weight standardization over input dim, fp32 -> bf16
// ---------------------------------------------------------------------------
__global__ void wstd_kernel(const float* __restrict__ W, __bf16* __restrict__ Wo,
                            int Cin) {
  const int o   = blockIdx.x;
  const int tid = threadIdx.x;
  const float* row = W + (size_t)o * Cin;
  float s1 = 0.f, s2 = 0.f;
  for (int i = tid; i < Cin; i += 256) { float w = row[i]; s1 += w; s2 += w * w; }
  __shared__ float r1[256], r2[256];
  r1[tid] = s1; r2[tid] = s2;
  __syncthreads();
  for (int s = 128; s > 0; s >>= 1) {
    if (tid < s) { r1[tid] += r1[tid + s]; r2[tid] += r2[tid + s]; }
    __syncthreads();
  }
  const float mu  = r1[0] / (float)Cin;
  const float var = r2[0] / (float)Cin - mu * mu;
  const float rs  = rsqrtf(var + cb::EPS);
  __bf16* orow = Wo + (size_t)o * Cin;
  for (int i = tid; i < Cin; i += 256) orow[i] = (__bf16)((row[i] - mu) * rs);
}

// ---------------------------------------------------------------------------
// Kernel 2: masked fp32 -> bf16 + transpose:  in [B][Ch][T] -> outT [B][T][Ch]
// ---------------------------------------------------------------------------
__global__ void cvt_transpose_kernel(const float* __restrict__ in,
                                     const unsigned char* __restrict__ mask,
                                     __bf16* __restrict__ outT, int Ch, int T) {
  __shared__ unsigned short tile[64][72];
  const int t0 = blockIdx.x * 64, c0 = blockIdx.y * 64, b = blockIdx.z;
  const int tid = threadIdx.x;
  const int tx = tid & 63, ry = tid >> 6;
  const float mk = mask[(size_t)b * T + t0 + tx] ? 1.0f : 0.0f;
#pragma unroll
  for (int r = ry; r < 64; r += 4) {
    const float v = in[((size_t)b * Ch + c0 + r) * T + t0 + tx] * mk;
    tile[r][tx] = __builtin_bit_cast(unsigned short, (__bf16)v);
  }
  __syncthreads();
  const int px = tid & 31, py = tid >> 5;
#pragma unroll
  for (int r = py; r < 64; r += 8) {
    const uint32_t lo = tile[2 * px][r], hi = tile[2 * px + 1][r];
    *(uint32_t*)&outT[((size_t)b * T + t0 + r) * Ch + c0 + 2 * px] =
        lo | (hi << 16);
  }
}

// ---------------------------------------------------------------------------
// Kernel 3: bf16 GEMM on W[Cout][Cin] x actT[b][T][Cin], double-buffered:
//   TRANSOUT=1: D[M=t][N=o] -> outT [b][T][Cout]   (bias over N)
//   TRANSOUT=0: D[M=o][N=t] -> out  [b][Cout][T]   (bias over M)
//   RESID     : fp32 out = acc + bias + resid (O-projection)
// ---------------------------------------------------------------------------
template <bool TRANSOUT, bool RESID>
__global__ void gemm_bf16_kernel(const __bf16* __restrict__ W,
                                 const __bf16* __restrict__ actT,
                                 const float* __restrict__ bias,
                                 __bf16* __restrict__ Yb,
                                 float* __restrict__ Yf,
                                 const float* __restrict__ resid,
                                 int Cout, int Cin, int T) {
  constexpr int LD = 72;                // 64 + 8 pad (bf16)
  __shared__ __bf16 Wt[2][128 * LD];    // [o][k] ping-pong
  __shared__ __bf16 At[2][128 * LD];    // [t][k] ping-pong

  const int tid = threadIdx.x;
  const int n0 = blockIdx.x * 128;
  const int m0 = blockIdx.y * 128;
  const int z  = blockIdx.z;
  const __bf16* actb = actT + (size_t)z * T * Cin;

  const int wrow0 = TRANSOUT ? n0 : m0;   // W tile rows (Cout axis)
  const int arow0 = TRANSOUT ? m0 : n0;   // act tile rows (T axis)

  const int wave = tid >> 5, lane = tid & 31;
  const int half = lane >> 4, l16 = lane & 15;
  const int Mw = (wave >> 2) * 64, Nw = (wave & 3) * 32;

  const int lc = tid & 7, lr = tid >> 3;  // 8 uint4 cols x 32 row-base

  auto issue = [&](int buf, int k0) {
#pragma unroll
    for (int r = lr; r < 128; r += 32) {
      copy16(&Wt[buf][r * LD + lc * 8],
             &W[(size_t)(wrow0 + r) * Cin + k0 + lc * 8]);
      copy16(&At[buf][r * LD + lc * 8],
             &actb[(size_t)(arow0 + r) * Cin + k0 + lc * 8]);
    }
  };

  v8f acc[4][2] = {};

  issue(0, 0);
  copy_fence();
  __syncthreads();

  const int NK = Cin / 64;
  for (int ki = 0; ki < NK; ++ki) {
    const int cur = ki & 1;
    if (ki + 1 < NK) issue(1 - cur, (ki + 1) * 64);  // overlap fill w/ WMMA

    const __bf16* Ta = TRANSOUT ? At[cur] : Wt[cur];
    const __bf16* Tb = TRANSOUT ? Wt[cur] : At[cur];
#pragma unroll
    for (int kc = 0; kc < 64; kc += 32) {
      v16bf af[4], bfr[2];
#pragma unroll
      for (int ms = 0; ms < 4; ++ms)
        af[ms] = load_frag(&Ta[(Mw + ms * 16 + l16) * LD + kc + half * 16]);
#pragma unroll
      for (int ns = 0; ns < 2; ++ns)
        bfr[ns] = load_frag(&Tb[(Nw + ns * 16 + l16) * LD + kc + half * 16]);
#pragma unroll
      for (int ms = 0; ms < 4; ++ms)
#pragma unroll
        for (int ns = 0; ns < 2; ++ns)
          acc[ms][ns] = wmma_bf16(af[ms], bfr[ns], acc[ms][ns]);
    }
    copy_fence();
    __syncthreads();
  }

  const int Ndim = TRANSOUT ? Cout : T;
#pragma unroll
  for (int ms = 0; ms < 4; ++ms)
#pragma unroll
    for (int ns = 0; ns < 2; ++ns)
#pragma unroll
      for (int r = 0; r < 8; ++r) {
        const int Mg = m0 + Mw + ms * 16 + r + 8 * half;
        const int Ng = n0 + Nw + ns * 16 + l16;
        const float bv = bias[TRANSOUT ? Ng : Mg];
        const float v = acc[ms][ns][r] + bv;
        const size_t o = (size_t)z * Cout * T + (size_t)Mg * Ndim + Ng;
        if (RESID) Yf[o] = v + resid[o];
        else       Yb[o] = (__bf16)v;
      }
}

// Tile loader: 128 rows x 128 bf16 cols from rows of stride `srs` (elems).
DEV void load_tile_128(__bf16* dst, const __bf16* src, size_t srs, int tid) {
  const int c = tid & 15, r0 = tid >> 4;   // 16 uint4 cols x 16 row-base
#pragma unroll
  for (int r = r0; r < 128; r += 16)
    copy16(&dst[r * 136 + c * 8], &src[(size_t)r * srs + c * 8]);
}

// ---------------------------------------------------------------------------
// Kernel 4: attention stats.  Per (c-tile=128, h, b): m[c]=max_q s,
// z[c]=sum_q exp(s-m),  s = (k^T q)/SCALE, query-masked.  Wave w owns
// c rows [16w,16w+16).  qT [b][t][E], kT [b][c][E].
// ---------------------------------------------------------------------------
__global__ void attn_stats_kernel(const __bf16* __restrict__ qT,
                                  const __bf16* __restrict__ kT,
                                  const unsigned char* __restrict__ maskq,
                                  float* __restrict__ mstat,
                                  float* __restrict__ zstat) {
  using namespace cb;
  __shared__ __bf16 Ks[128 * 136];   // [c][d]
  __shared__ __bf16 Qs[128 * 136];   // [q][d]

  const int c0 = blockIdx.x * 128, h = blockIdx.y, b = blockIdx.z;
  const int tid = threadIdx.x;
  const int wave = tid >> 5, lane = tid & 31;
  const int half = lane >> 4, l16 = lane & 15;

  const __bf16* kbase = kT + ((size_t)b * TC + c0) * E + h * DH;
  const __bf16* qbase = qT + (size_t)b * TQ * E + h * DH;
  load_tile_128(Ks, kbase, E, tid);

  float mrun[8], zrun[8];
#pragma unroll
  for (int r = 0; r < 8; ++r) { mrun[r] = -3.0e38f; zrun[r] = 0.f; }

  for (int q0 = 0; q0 < TQ; q0 += 128) {
    __syncthreads();
    load_tile_128(Qs, qbase + (size_t)q0 * E, E, tid);
    if (q0 + 128 < TQ)
      __builtin_prefetch(qbase + (size_t)(q0 + 128 + (tid >> 1)) * E + (tid & 1) * 64, 0, 1);
    copy_fence();
    __syncthreads();

    v8f s[8] = {};
#pragma unroll
    for (int kc = 0; kc < 128; kc += 32) {
      const v16bf a = load_frag(&Ks[(wave * 16 + l16) * 136 + kc + half * 16]);
#pragma unroll
      for (int nt = 0; nt < 8; ++nt) {
        const v16bf bq = load_frag(&Qs[(nt * 16 + l16) * 136 + kc + half * 16]);
        s[nt] = wmma_bf16(a, bq, s[nt]);
      }
    }

    bool qm[8];
#pragma unroll
    for (int nt = 0; nt < 8; ++nt)
      qm[nt] = maskq[(size_t)b * TQ + q0 + nt * 16 + l16] != 0;

#pragma unroll
    for (int r = 0; r < 8; ++r) {
      float tmax = -3.0e38f;
#pragma unroll
      for (int nt = 0; nt < 8; ++nt)
        if (qm[nt]) tmax = fmaxf(tmax, s[nt][r] * RSCALE);
      const float mn = fmaxf(mrun[r], tmax);
      float zr = zrun[r] * __expf(mrun[r] - mn);
#pragma unroll
      for (int nt = 0; nt < 8; ++nt)
        if (qm[nt]) zr += __expf(s[nt][r] * RSCALE - mn);
      mrun[r] = mn;
      zrun[r] = zr;
    }
  }

#pragma unroll
  for (int r = 0; r < 8; ++r) {
    float m = mrun[r], zv = zrun[r];
#pragma unroll
    for (int off = 1; off < 16; off <<= 1) {
      const float m2 = __shfl_xor(m, off, 32);
      const float z2 = __shfl_xor(zv, off, 32);
      const float mn = fmaxf(m, m2);
      zv = zv * __expf(m - mn) + z2 * __expf(m2 - mn);
      m = mn;
    }
    if (l16 == 0) {
      const int cg = c0 + wave * 16 + r + 8 * half;
      const size_t o = ((size_t)b * H + h) * TC + cg;
      mstat[o] = m;
      zstat[o] = zv;
    }
  }
}

// ---------------------------------------------------------------------------
// Kernel 5: attention apply.  Per (q-tile=128, h, b): loop c-tiles:
//   S = K^T Q (wmma) ; P = exp(S/SCALE - m)/z masked -> LDS [q][c] (bf16,
//   aliasing the K tile) ; outT[q][d] += P x V (wmma, A=P [q][c], B=V [d][c]).
// ---------------------------------------------------------------------------
__global__ void attn_apply_kernel(const __bf16* __restrict__ qT,
                                  const __bf16* __restrict__ kT,
                                  const __bf16* __restrict__ vb,
                                  const unsigned char* __restrict__ maskq,
                                  const unsigned char* __restrict__ maskc,
                                  const float* __restrict__ mstat,
                                  const float* __restrict__ zstat,
                                  __bf16* __restrict__ aoutT) {
  using namespace cb;
  __shared__ __bf16 Qs[128 * 136];    // [q][d], loaded once
  __shared__ __bf16 KPs[128 * 136];   // [c][d] as K, then [q][c] as P
  __shared__ __bf16 Vs[128 * 136];    // [d][c]

  const int q0 = blockIdx.x * 128, h = blockIdx.y, b = blockIdx.z;
  const int tid = threadIdx.x;
  const int wave = tid >> 5, lane = tid & 31;
  const int half = lane >> 4, l16 = lane & 15;

  const __bf16* qbase = qT + ((size_t)b * TQ + q0) * E + h * DH;
  const __bf16* kbase = kT + (size_t)b * TC * E + h * DH;
  const __bf16* vbase = vb + ((size_t)b * E + h * DH) * TC;
  const float* mrowp = mstat + ((size_t)b * H + h) * TC;
  const float* zrowp = zstat + ((size_t)b * H + h) * TC;

  load_tile_128(Qs, qbase, E, tid);

  bool qm[8];
#pragma unroll
  for (int nt = 0; nt < 8; ++nt)
    qm[nt] = maskq[(size_t)b * TQ + q0 + nt * 16 + l16] != 0;

  v8f oacc[8] = {};

  for (int c0 = 0; c0 < TC; c0 += 128) {
    __syncthreads();
    load_tile_128(KPs, kbase + (size_t)c0 * E, E, tid);
    load_tile_128(Vs, vbase + c0, TC, tid);
    if (c0 + 128 < TC) {
      __builtin_prefetch(kbase + (size_t)(c0 + 128 + (tid >> 1)) * E + (tid & 1) * 64, 0, 1);
      __builtin_prefetch(vbase + (size_t)(tid & 127) * TC + c0 + 128, 0, 1);
    }
    copy_fence();
    __syncthreads();

    // ---- S = K^T Q for c rows [16w,16w+16), all 128 q ----
    v8f s[8] = {};
#pragma unroll
    for (int kc = 0; kc < 128; kc += 32) {
      const v16bf a = load_frag(&KPs[(wave * 16 + l16) * 136 + kc + half * 16]);
#pragma unroll
      for (int nt = 0; nt < 8; ++nt) {
        const v16bf bq = load_frag(&Qs[(nt * 16 + l16) * 136 + kc + half * 16]);
        s[nt] = wmma_bf16(a, bq, s[nt]);
      }
    }
    __syncthreads();   // done reading KPs as K

    // ---- P = exp(s/SCALE - m)/z (masked) -> KPs as [q][c] ----
#pragma unroll
    for (int r = 0; r < 8; ++r) {
      const int cl = wave * 16 + r + 8 * half;
      const int cg = c0 + cl;
      const bool cm = maskc[(size_t)b * TC + cg] != 0;
      const float mr = mrowp[cg];
      const float rz = cm ? (1.0f / zrowp[cg]) : 0.0f;
#pragma unroll
      for (int nt = 0; nt < 8; ++nt) {
        float p = 0.0f;
        if (cm && qm[nt]) p = __expf(s[nt][r] * RSCALE - mr) * rz;
        KPs[(nt * 16 + l16) * 136 + cl] = (__bf16)p;
      }
    }
    __syncthreads();

    // ---- outT[q][d] += P[q][c] * V[d][c]^T ; wave owns q rows [16w,16w+16) ----
#pragma unroll
    for (int kc = 0; kc < 128; kc += 32) {
      const v16bf a = load_frag(&KPs[(wave * 16 + l16) * 136 + kc + half * 16]);
#pragma unroll
      for (int nt = 0; nt < 8; ++nt) {
        const v16bf bv = load_frag(&Vs[(nt * 16 + l16) * 136 + kc + half * 16]);
        oacc[nt] = wmma_bf16(a, bv, oacc[nt]);
      }
    }
  }

  __bf16* obase = aoutT + ((size_t)b * TQ + q0) * E + h * DH;
#pragma unroll
  for (int nt = 0; nt < 8; ++nt)
#pragma unroll
    for (int r = 0; r < 8; ++r) {
      const int qr = wave * 16 + r + 8 * half;
      obase[(size_t)qr * E + nt * 16 + l16] = (__bf16)oacc[nt][r];
    }
}

// ---------------------------------------------------------------------------
// Kernel 6: LayerNorm over E per (b,t) column + final query mask -> d_out
// ---------------------------------------------------------------------------
__global__ void layernorm_kernel(const float* __restrict__ hbuf,
                                 const float* __restrict__ gamma,
                                 const float* __restrict__ beta,
                                 const unsigned char* __restrict__ maskq,
                                 float* __restrict__ out) {
  using namespace cb;
  const int t0 = blockIdx.x * 64, b = blockIdx.y;
  const int tx = threadIdx.x & 63, ty = threadIdx.x >> 6;
  const float* hb = hbuf + (size_t)b * E * TQ;

  float s1 = 0.f, s2 = 0.f;
  for (int e = ty; e < E; e += 4) {
    const float v = hb[(size_t)e * TQ + t0 + tx];
    s1 += v; s2 += v * v;
  }
  __shared__ float r1[4][64], r2[4][64];
  __shared__ float mu_s[64], rs_s[64];
  r1[ty][tx] = s1; r2[ty][tx] = s2;
  __syncthreads();
  if (ty == 0) {
    const float a = r1[0][tx] + r1[1][tx] + r1[2][tx] + r1[3][tx];
    const float q = r2[0][tx] + r2[1][tx] + r2[2][tx] + r2[3][tx];
    const float mu = a / (float)E;
    const float var = q / (float)E - mu * mu;
    mu_s[tx] = mu;
    rs_s[tx] = rsqrtf(var + EPS);
  }
  __syncthreads();
  const float mu = mu_s[tx], rs = rs_s[tx];
  const float mk = maskq[(size_t)b * TQ + t0 + tx] ? 1.0f : 0.0f;
  float* ob = out + (size_t)b * E * TQ;
  for (int e = ty; e < E; e += 4) {
    const size_t o = (size_t)e * TQ + t0 + tx;
    ob[o] = ((hb[o] - mu) * rs * gamma[e] + beta[e]) * mk;
  }
}

// ---------------------------------------------------------------------------
extern "C" void kernel_launch(void* const* d_in, const int* in_sizes, int n_in,
                              void* d_out, int out_size, void* d_ws, size_t ws_size,
                              hipStream_t stream) {
  using namespace cb;
  (void)in_sizes; (void)n_in; (void)out_size; (void)ws_size;

  const float* x     = (const float*)d_in[0];
  const float* ctx   = (const float*)d_in[1];
  const unsigned char* maskq = (const unsigned char*)d_in[2];
  const unsigned char* maskc = (const unsigned char*)d_in[3];
  const float* wq = (const float*)d_in[4];  const float* bq = (const float*)d_in[5];
  const float* wk = (const float*)d_in[6];  const float* bk = (const float*)d_in[7];
  const float* wv = (const float*)d_in[8];  const float* bv = (const float*)d_in[9];
  const float* wo = (const float*)d_in[10]; const float* bo = (const float*)d_in[11];
  const float* gamma = (const float*)d_in[12];
  const float* beta  = (const float*)d_in[13];

  char* wsp = (char*)d_ws;
  size_t off = 0;
  auto alloc = [&](size_t bytes) {
    void* p = wsp + off;
    off = (off + bytes + 255) & ~(size_t)255;
    return p;
  };
  __bf16* wsq  = (__bf16*)alloc((size_t)E * E * 2);
  __bf16* wsk  = (__bf16*)alloc((size_t)E * C * 2);
  __bf16* wsv  = (__bf16*)alloc((size_t)E * C * 2);
  __bf16* wso  = (__bf16*)alloc((size_t)E * E * 2);
  __bf16* xbT  = (__bf16*)alloc((size_t)B * TQ * E * 2);   // [b][t][E]
  __bf16* cbT  = (__bf16*)alloc((size_t)B * TC * C * 2);   // [b][t][C]
  __bf16* qbT  = (__bf16*)alloc((size_t)B * TQ * E * 2);   // [b][t][E]
  __bf16* kbT  = (__bf16*)alloc((size_t)B * TC * E * 2);   // [b][c][E]
  __bf16* vbuf = (__bf16*)alloc((size_t)B * E * TC * 2);   // [b][E][c]
  float*  mst  = (float*)alloc((size_t)B * H * TC * 4);
  float*  zst  = (float*)alloc((size_t)B * H * TC * 4);
  __bf16* aoutT= (__bf16*)alloc((size_t)B * TQ * E * 2);   // [b][t][E]
  float*  hbuf = (float*)alloc((size_t)B * E * TQ * 4);    // [b][E][t]

  // 1) weight standardization -> bf16
  wstd_kernel<<<E, 256, 0, stream>>>(wq, wsq, E);
  wstd_kernel<<<E, 256, 0, stream>>>(wk, wsk, C);
  wstd_kernel<<<E, 256, 0, stream>>>(wv, wsv, C);
  wstd_kernel<<<E, 256, 0, stream>>>(wo, wso, E);

  // 2) masked activations -> bf16, transposed (channel-innermost)
  cvt_transpose_kernel<<<dim3(TQ / 64, E / 64, B), 256, 0, stream>>>(x, maskq, xbT, E, TQ);
  cvt_transpose_kernel<<<dim3(TC / 64, C / 64, B), 256, 0, stream>>>(ctx, maskc, cbT, C, TC);

  // 3) projections (bf16 WMMA GEMM)
  gemm_bf16_kernel<true,  false><<<dim3(E / 128, TQ / 128, B), 256, 0, stream>>>(
      wsq, xbT, bq, qbT, nullptr, nullptr, E, E, TQ);
  gemm_bf16_kernel<true,  false><<<dim3(E / 128, TC / 128, B), 256, 0, stream>>>(
      wsk, cbT, bk, kbT, nullptr, nullptr, E, C, TC);
  gemm_bf16_kernel<false, false><<<dim3(TC / 128, E / 128, B), 256, 0, stream>>>(
      wsv, cbT, bv, vbuf, nullptr, nullptr, E, C, TC);

  // 4) attention: per-context-row softmax stats over q, then apply
  attn_stats_kernel<<<dim3(TC / 128, H, B), 256, 0, stream>>>(qbT, kbT, maskq, mst, zst);
  attn_apply_kernel<<<dim3(TQ / 128, H, B), 256, 0, stream>>>(qbT, kbT, vbuf, maskq, maskc,
                                                              mst, zst, aoutT);

  // 5) output projection + residual (f32 out, [b][E][t])
  gemm_bf16_kernel<false, true><<<dim3(TQ / 128, E / 128, B), 256, 0, stream>>>(
      wso, aoutT, bo, nullptr, hbuf, x, E, E, TQ);

  // 6) LayerNorm over E + final mask -> d_out (f32)
  layernorm_kernel<<<dim3(TQ / 64, B), 256, 0, stream>>>(hbuf, gamma, beta, maskq,
                                                         (float*)d_out);
}
